// FMG_6717328851807
// MI455X (gfx1250) — matrix operand order repaired and checked by reading, not compiled, not run on hardware
//
#include <hip/hip_runtime.h>

// MI455X / gfx1250 fused memory-token attention block, f16 WMMA pipeline.
// q = Wq@x (1x1 conv), k/v = Ft@W^T, per-head attention over 64 memory tokens,
// output projection Wp + bias + residual. All GEMMs: v_wmma_f32_16x16x32_f16
// with f32 accumulation. Large GEMMs use double-buffered LDS staging so
// global loads for tile k+1 overlap the WMMA work on tile k.

typedef __attribute__((ext_vector_type(16))) _Float16 v16h;
typedef __attribute__((ext_vector_type(8)))  float    v8f;

#define B_     8
#define C_     512
#define N_     4096
#define KTOK   64
#define MEM_   512
#define HEADS_ 8
#define DHEAD  64

// ---------------- helpers ----------------------------------------------------

__device__ __forceinline__ unsigned pack2(float a, float b) {
  union { _Float16 h[2]; unsigned u; } u;
  u.h[0] = (_Float16)a;
  u.h[1] = (_Float16)b;
  return u.u;
}

// A-matrix 16x32 (MxK) f16 fragment (layout per CDNA5 ISA):
// lanes 0-15 -> M=lane, VGPR v<4: K=2v,2v+1; v>=4: K=16+2(v-4); lanes 16-31
// same M with K offset +8 inside each 16-group. Pairs are contiguous -> the
// compiler merges these into ds_load_b128.
__device__ __forceinline__ v16h frag_a(const _Float16* s, int stride, int m0,
                                       int k0, int lane) {
  const int m = lane & 15, kh = lane >> 4;
  const _Float16* row = s + (m0 + m) * stride + k0;
  v16h f;
#pragma unroll
  for (int v = 0; v < 8; ++v) {
    const int kb = ((v >> 2) << 4) | (kh << 3) | ((v & 3) << 1);
    f[2 * v]     = row[kb];
    f[2 * v + 1] = row[kb + 1];
  }
  return f;
}

// B-matrix 32x16 (KxN) f16 fragment sourced from a B^T tile: element (k,n) at
// s[n*stride + k]. Lanes 0-15 -> N=lane, same K order as frag_a; lanes 16-31
// same N with K offset +8 inside each 16-group.
__device__ __forceinline__ v16h frag_bt(const _Float16* s, int stride, int n0,
                                        int k0, int lane) {
  const int n = lane & 15, kh = lane >> 4;
  const _Float16* col = s + (n0 + n) * stride + k0;
  v16h f;
#pragma unroll
  for (int v = 0; v < 8; ++v) {
    const int kb = ((v >> 2) << 4) | (kh << 3) | ((v & 3) << 1);
    f[2 * v]     = col[kb];
    f[2 * v + 1] = col[kb + 1];
  }
  return f;
}

__device__ __forceinline__ v8f wmma16(v16h a, v16h b, v8f c) {
  return __builtin_amdgcn_wmma_f32_16x16x32_f16(false, a, false, b, (short)0, c,
                                                false, false);
}

// ---------------- k/v projection: [B,64tok,MEM] @ W^T -> [B,h,64,64] f16 -----
__global__ void __launch_bounds__(128)
kv_kernel(const float* __restrict__ Ft, const float* __restrict__ Wk,
          const float* __restrict__ Wv, _Float16* __restrict__ k_ws,
          _Float16* __restrict__ v_ws) {
  const int b = blockIdx.z;
  const int head = blockIdx.x;          // 64-channel tile == one head
  const float* W = blockIdx.y ? Wv : Wk;
  _Float16* outp = blockIdx.y ? v_ws : k_ws;
  const int n0g = head * 64;
  const int tid = threadIdx.x, lane = tid & 31, wave = tid >> 5;
  const int lh = lane >> 4, ln = lane & 15;

  __shared__ __align__(16) _Float16 As[64 * 40];   // [token][k]
  __shared__ __align__(16) _Float16 Bs[64 * 40];   // B^T: [chan][k]

  v8f z = {};
  v8f acc[4] = {z, z, z, z};

  for (int k0 = 0; k0 < MEM_; k0 += 32) {
#pragma unroll
    for (int i = 0; i < 8; ++i) {
      const int idx = i * 128 + tid;
      const int r = idx >> 4, p = (idx & 15) << 1;
      const float2 fa = *(const float2*)&Ft[((size_t)b * KTOK + r) * MEM_ + k0 + p];
      *(unsigned*)&As[r * 40 + p] = pack2(fa.x, fa.y);
      const float2 fb = *(const float2*)&W[(size_t)(n0g + r) * MEM_ + k0 + p];
      *(unsigned*)&Bs[r * 40 + p] = pack2(fb.x, fb.y);
    }
    __syncthreads();
    const v16h a = frag_a(As, 40, wave * 16, 0, lane);
#pragma unroll
    for (int nt = 0; nt < 4; ++nt) {
      const v16h bf = frag_bt(Bs, 40, nt * 16, 0, lane);
      acc[nt] = wmma16(a, bf, acc[nt]);
    }
    __syncthreads();
  }

#pragma unroll
  for (int nt = 0; nt < 4; ++nt)
#pragma unroll
    for (int r = 0; r < 8; ++r) {
      const int t = wave * 16 + r + 8 * lh;      // token
      const int c = nt * 16 + ln;                // channel within head
      outp[((size_t)(b * HEADS_ + head) * KTOK + t) * DHEAD + c] =
          (_Float16)acc[nt][r];
    }
}

// ---------------- q projection: Wq @ x -> q_ws [B,h,N,64] f16 ----------------
// 256 threads, 128x128 tile, double-buffered staging.
__device__ __forceinline__ void stage_qproj(_Float16* As, _Float16* Bs,
                                            const float* __restrict__ Wq,
                                            const float* __restrict__ xb,
                                            int m0, int n0, int k0, int tid) {
#pragma unroll
  for (int i = 0; i < 8; ++i) {
    const int idx = i * 256 + tid;
    const int r = idx >> 4, p = (idx & 15) << 1;
    const float2 w = *(const float2*)&Wq[(size_t)(m0 + r) * C_ + k0 + p];
    *(unsigned*)&As[r * 40 + p] = pack2(w.x, w.y);
  }
#pragma unroll
  for (int i = 0; i < 8; ++i) {
    const int idx = i * 256 + tid;
    const int tn = idx & 127, kp = idx >> 7;     // kp in 0..15
    const float b0 = xb[(size_t)(k0 + 2 * kp) * N_ + n0 + tn];
    const float b1 = xb[(size_t)(k0 + 2 * kp + 1) * N_ + n0 + tn];
    *(unsigned*)&Bs[tn * 40 + 2 * kp] = pack2(b0, b1);
  }
}

__global__ void __launch_bounds__(256)
qproj_kernel(const float* __restrict__ x, const float* __restrict__ Wq,
             _Float16* __restrict__ q_ws) {
  const int b = blockIdx.z;
  const int m0 = blockIdx.y * 128;      // output channel tile (2 heads)
  const int n0 = blockIdx.x * 128;      // token tile
  const int tid = threadIdx.x, lane = tid & 31, wave = tid >> 5;
  const int wm = wave >> 1, wn = wave & 1;
  const int lh = lane >> 4, ln = lane & 15;

  // 2 staging buffers of (A[128x40] | B[128x40]); also reused as the 128x136
  // output-transpose buffer (17408 <= 20480 halves).
  __shared__ __align__(16) _Float16 sh[2 * 128 * 80];

  v8f z = {};
  v8f acc[2][4] = {{z, z, z, z}, {z, z, z, z}};
  const float* xb = x + (size_t)b * C_ * N_;

  stage_qproj(sh, sh + 5120, Wq, xb, m0, n0, 0, tid);
  for (int ks = 0; ks < 16; ++ks) {
    __syncthreads();
    _Float16* cur = sh + (ks & 1) * 10240;
    if (ks + 1 < 16)
      stage_qproj(sh + ((ks + 1) & 1) * 10240,
                  sh + ((ks + 1) & 1) * 10240 + 5120, Wq, xb, m0, n0,
                  (ks + 1) * 32, tid);
    const v16h a0 = frag_a(cur, 40, wm * 32, 0, lane);
    const v16h a1 = frag_a(cur, 40, wm * 32 + 16, 0, lane);
    _Float16* curB = cur + 5120;
#pragma unroll
    for (int nt = 0; nt < 4; ++nt) {
      const v16h bf = frag_bt(curB, 40, wn * 64 + nt * 16, 0, lane);
      acc[0][nt] = wmma16(a0, bf, acc[0][nt]);
      acc[1][nt] = wmma16(a1, bf, acc[1][nt]);
    }
  }
  __syncthreads();

  // transpose tile in LDS -> sh[token][chan] (stride 136, 272B = 16-aligned)
#pragma unroll
  for (int mi = 0; mi < 2; ++mi)
#pragma unroll
    for (int nt = 0; nt < 4; ++nt)
#pragma unroll
      for (int r = 0; r < 8; ++r) {
        const int tl = wn * 64 + nt * 16 + ln;
        const int cl = wm * 32 + mi * 16 + r + 8 * lh;
        sh[tl * 136 + cl] = (_Float16)acc[mi][nt][r];
      }
  __syncthreads();
  // coalesced b128 rows into q_ws[b][h][token][d]
#pragma unroll
  for (int i = 0; i < 8; ++i) {
    const int idx = i * 256 + tid;
    const int tl = idx >> 4, ol = (idx & 15) << 3;    // 16 chunks of 8 halves
    const int h = (m0 >> 6) + (ol >> 6);
    const int d = ol & 63;
    *(uint4*)&q_ws[((size_t)(b * HEADS_ + h) * N_ + n0 + tl) * DHEAD + d] =
        *(const uint4*)&sh[tl * 136 + ol];
  }
}

// ---------------- fused attention: scores -> softmax -> attn@v ---------------
// ao_ws layout: [B][N][C] f16 (token-major) for coalesced final-GEMM staging.
__global__ void __launch_bounds__(128)
attn_kernel(const _Float16* __restrict__ q_ws, const _Float16* __restrict__ k_ws,
            const _Float16* __restrict__ v_ws, _Float16* __restrict__ ao_ws) {
  const int b = blockIdx.z, hh = blockIdx.y;
  const int t0 = blockIdx.x * 64;
  const int tid = threadIdx.x, lane = tid & 31, wave = tid >> 5;
  const int lh = lane >> 4, ln = lane & 15;

  __shared__ __align__(16) _Float16 qs[64 * 72];   // [token][d]
  __shared__ __align__(16) _Float16 ks[64 * 72];   // [keytok][d]  (B^T of q@k^T)
  __shared__ __align__(16) _Float16 vT[64 * 72];   // [d][keytok]  (B^T of p@v)
  __shared__ __align__(16) float    Sf[64 * 68];   // scores f32
  __shared__ __align__(16) _Float16 ps[64 * 72];   // probs f16

  const _Float16* qb = q_ws + ((size_t)(b * HEADS_ + hh) * N_ + t0) * DHEAD;
  const _Float16* kb = k_ws + (size_t)(b * HEADS_ + hh) * KTOK * DHEAD;
  const _Float16* vb = v_ws + (size_t)(b * HEADS_ + hh) * KTOK * DHEAD;

  // 64x64 tiles: 512 16B-chunks each -> 4 per thread; v is scattered (transpose)
#pragma unroll
  for (int i = 0; i < 4; ++i) {
    const int idx = i * 128 + tid;
    const int r = idx >> 3, cc = (idx & 7) << 3;
    *(uint4*)&qs[r * 72 + cc] = *(const uint4*)&qb[r * 64 + cc];
    *(uint4*)&ks[r * 72 + cc] = *(const uint4*)&kb[r * 64 + cc];
    const uint4 vv = *(const uint4*)&vb[r * 64 + cc];
    const _Float16* hp = (const _Float16*)&vv;
#pragma unroll
    for (int j = 0; j < 8; ++j) vT[(cc + j) * 72 + r] = hp[j];
  }
  __syncthreads();

  // S = (q k^T) / sqrt(64); each wave: 16 query rows x 64 keys
  v8f z = {};
  v8f s[4] = {z, z, z, z};
#pragma unroll
  for (int kk = 0; kk < 2; ++kk) {
    const v16h a = frag_a(qs, 72, wave * 16, kk * 32, lane);
#pragma unroll
    for (int nt = 0; nt < 4; ++nt) {
      const v16h bf = frag_bt(ks, 72, nt * 16, kk * 32, lane);
      s[nt] = wmma16(a, bf, s[nt]);
    }
  }
#pragma unroll
  for (int nt = 0; nt < 4; ++nt)
#pragma unroll
    for (int r = 0; r < 8; ++r)
      Sf[(wave * 16 + r + 8 * lh) * 68 + nt * 16 + ln] = s[nt][r] * 0.125f;
  __syncthreads();

  // softmax over 64 keys: one full row per thread (threads 0..63)
  if (tid < 64) {
    float* row = &Sf[tid * 68];
    float mx = row[0];
#pragma unroll
    for (int j = 1; j < 64; ++j) mx = fmaxf(mx, row[j]);
    float sum = 0.f;
#pragma unroll
    for (int j = 0; j < 64; ++j) {
      const float e = __expf(row[j] - mx);
      row[j] = e;
      sum += e;
    }
    const float inv = 1.f / sum;
#pragma unroll
    for (int j = 0; j < 32; ++j)
      *(unsigned*)&ps[tid * 72 + 2 * j] =
          pack2(row[2 * j] * inv, row[2 * j + 1] * inv);
  }
  __syncthreads();

  // out = P @ v  (inner dim = 64 key tokens)
  v8f o[4] = {z, z, z, z};
#pragma unroll
  for (int kk = 0; kk < 2; ++kk) {
    const v16h a = frag_a(ps, 72, wave * 16, kk * 32, lane);
#pragma unroll
    for (int nt = 0; nt < 4; ++nt) {
      const v16h bf = frag_bt(vT, 72, nt * 16, kk * 32, lane);
      o[nt] = wmma16(a, bf, o[nt]);
    }
  }
#pragma unroll
  for (int nt = 0; nt < 4; ++nt)
#pragma unroll
    for (int r = 0; r < 8; ++r) {
      const int t = t0 + wave * 16 + r + 8 * lh;
      const int ch = hh * 64 + nt * 16 + ln;
      ao_ws[((size_t)b * N_ + t) * C_ + ch] = (_Float16)o[nt][r];
    }
}

// ---------------- output projection + bias + residual ------------------------
// 256 threads, 128x128 tile, double-buffered staging; ao staged as f16 b128.
__device__ __forceinline__ void stage_proj(_Float16* As, _Float16* Bs,
                                           const float* __restrict__ Wp,
                                           const _Float16* __restrict__ aob,
                                           int m0, int n0, int k0, int tid) {
#pragma unroll
  for (int i = 0; i < 8; ++i) {
    const int idx = i * 256 + tid;
    const int r = idx >> 4, p = (idx & 15) << 1;
    const float2 w = *(const float2*)&Wp[(size_t)(m0 + r) * C_ + k0 + p];
    *(unsigned*)&As[r * 40 + p] = pack2(w.x, w.y);
  }
#pragma unroll
  for (int i = 0; i < 2; ++i) {
    const int idx = i * 256 + tid;
    const int r = idx >> 2, cc = (idx & 3) << 3;
    *(uint4*)&Bs[r * 40 + cc] =
        *(const uint4*)&aob[(size_t)(n0 + r) * C_ + k0 + cc];
  }
}

__global__ void __launch_bounds__(256)
proj_kernel(const _Float16* __restrict__ ao, const float* __restrict__ Wp,
            const float* __restrict__ bp, const float* __restrict__ x,
            float* __restrict__ out) {
  const int b = blockIdx.z;
  const int m0 = blockIdx.y * 128;      // output channel tile
  const int n0 = blockIdx.x * 128;      // token tile
  const int tid = threadIdx.x, lane = tid & 31, wave = tid >> 5;
  const int wm = wave >> 1, wn = wave & 1;
  const int lh = lane >> 4, ln = lane & 15;

  __shared__ __align__(16) _Float16 sh[2 * 128 * 80];  // 2x (A|B) buffers

  v8f z = {};
  v8f acc[2][4] = {{z, z, z, z}, {z, z, z, z}};
  const _Float16* aob = ao + (size_t)b * N_ * C_;

  stage_proj(sh, sh + 5120, Wp, aob, m0, n0, 0, tid);
  for (int ks = 0; ks < 16; ++ks) {
    __syncthreads();
    _Float16* cur = sh + (ks & 1) * 10240;
    if (ks + 1 < 16)
      stage_proj(sh + ((ks + 1) & 1) * 10240,
                 sh + ((ks + 1) & 1) * 10240 + 5120, Wp, aob, m0, n0,
                 (ks + 1) * 32, tid);
    const v16h a0 = frag_a(cur, 40, wm * 32, 0, lane);
    const v16h a1 = frag_a(cur, 40, wm * 32 + 16, 0, lane);
    _Float16* curB = cur + 5120;
#pragma unroll
    for (int nt = 0; nt < 4; ++nt) {
      const v16h bf = frag_bt(curB, 40, wn * 64 + nt * 16, 0, lane);
      acc[0][nt] = wmma16(a0, bf, acc[0][nt]);
      acc[1][nt] = wmma16(a1, bf, acc[1][nt]);
    }
  }

#pragma unroll
  for (int mi = 0; mi < 2; ++mi)
#pragma unroll
    for (int nt = 0; nt < 4; ++nt)
#pragma unroll
      for (int r = 0; r < 8; ++r) {
        const int o = m0 + wm * 32 + mi * 16 + r + 8 * lh;
        const int t = n0 + wn * 64 + nt * 16 + ln;
        const size_t gi = ((size_t)b * C_ + o) * N_ + t;
        out[gi] = acc[mi][nt][r] + bp[o] + x[gi];
      }
}

// ---------------- host launch ------------------------------------------------
extern "C" void kernel_launch(void* const* d_in, const int* in_sizes, int n_in,
                              void* d_out, int out_size, void* d_ws,
                              size_t ws_size, hipStream_t stream) {
  (void)in_sizes; (void)n_in; (void)out_size; (void)ws_size;
  const float* x  = (const float*)d_in[0];
  const float* Ft = (const float*)d_in[1];
  const float* Wq = (const float*)d_in[2];
  const float* Wk = (const float*)d_in[3];
  const float* Wv = (const float*)d_in[4];
  const float* Wp = (const float*)d_in[5];
  const float* bp = (const float*)d_in[6];
  float* out = (float*)d_out;

  char* ws = (char*)d_ws;
  _Float16* q_ws  = (_Float16*)(ws);                       // 33,554,432 B
  _Float16* ao_ws = (_Float16*)(ws + (size_t)33554432);    // 33,554,432 B
  _Float16* k_ws  = (_Float16*)(ws + (size_t)67108864);    //    524,288 B
  _Float16* v_ws  = (_Float16*)(ws + (size_t)67633152);    //    524,288 B

  kv_kernel<<<dim3(HEADS_, 2, B_), 128, 0, stream>>>(Ft, Wk, Wv, k_ws, v_ws);
  qproj_kernel<<<dim3(N_ / 128, C_ / 128, B_), 256, 0, stream>>>(x, Wq, q_ws);
  attn_kernel<<<dim3(N_ / 64, HEADS_, B_), 128, 0, stream>>>(q_ws, k_ws, v_ws,
                                                             ao_ws);
  proj_kernel<<<dim3(N_ / 128, C_ / 128, B_), 256, 0, stream>>>(ao_ws, Wp, bp,
                                                                x, out);
}
